// Attention_44358422233312
// MI455X (gfx1250) — matrix operand order repaired and checked by reading, not compiled, not run on hardware
//
#include <hip/hip_runtime.h>

// MI455X / gfx1250: wave32, WMMA f32_16x16x32_f16 (f32 accumulate).
// Pipeline: memset(pad) -> pack W to f16 B-layout -> fused qkv proj + LN
// (x read once; q/k/vT stored packed f16) -> flash attention with Gaussian bias.
// Round 4: A-tile loads made unconditional (invalid lanes clamp to row 0 and
// their outputs are never stored), so x is read with b128 vector loads instead
// of exec-masked scalar loads.

typedef __attribute__((ext_vector_type(16))) _Float16 v16h;
typedef __attribute__((ext_vector_type(8)))  _Float16 v8h;
typedef __attribute__((ext_vector_type(8)))  float    v8f;
typedef __attribute__((ext_vector_type(4)))  float    v4f;

#define T_TOK   1025               // GRID*GRID + 1
#define NTILE   65                 // ceil(1025/16)
#define TPAD    1040               // NTILE*16   (q rows padded)
#define KPAD    1056               // 33*32      (k rows / vT cols padded)
#define DH      64
#define DMODEL  768
#define NPATCH  1024
#define SCALE_INV 0.125f
#define NKC     33                 // key chunks of 32

__device__ __forceinline__ float red_max16(float v) {
    #pragma unroll
    for (int m = 1; m < 16; m <<= 1) v = fmaxf(v, __shfl_xor(v, m, 32));
    return v;
}
__device__ __forceinline__ float red_sum16(float v) {
    #pragma unroll
    for (int m = 1; m < 16; m <<= 1) v += __shfl_xor(v, m, 32);
    return v;
}
__device__ __forceinline__ v16h cat8(v8h a, v8h b) {
    return __builtin_shufflevector(a, b, 0,1,2,3,4,5,6,7,8,9,10,11,12,13,14,15);
}

// ---------------------------------------------------------------------------
// Pack W (768x64 fp32, row-major) into f16 WMMA B-lane layout:
//   WP[((kc*4+nt)*32 + lane)*16 + j] = W[(kc*32 + (lane>>4)*16 + j)*64 + nt*16+(lane&15)]
// One-time strided gather; hot loops then do pure 32B contiguous loads.
// ---------------------------------------------------------------------------
__global__ __launch_bounds__(32) void pack_w_kernel(
    const float* __restrict__ Wq, const float* __restrict__ Wk, const float* __restrict__ Wv,
    _Float16* __restrict__ WqP, _Float16* __restrict__ WkP, _Float16* __restrict__ WvP)
{
    const int lane = threadIdx.x;
    const int tile = blockIdx.x % 96;          // kc*4 + nt  (24 kc, 4 nt)
    const int p    = blockIdx.x / 96;
    const int kc   = tile >> 2, nt = tile & 3;
    const float*  W  = (p == 0) ? Wq  : (p == 1) ? Wk  : Wv;
    _Float16*     WP = (p == 0) ? WqP : (p == 1) ? WkP : WvP;
    const int n  = nt * 16 + (lane & 15);
    const int kb = kc * 32 + (lane >> 4) * 16;
    _Float16* dst = WP + ((size_t)tile * 32 + lane) * 16;
    #pragma unroll
    for (int j = 0; j < 16; ++j)
        dst[j] = (_Float16)W[(size_t)(kb + j) * DH + n];
}

// A-tile loader: fp32 x row -> f16, 16-bit A layout (two 8-elt runs).
// UNCONDITIONAL: caller clamps the row pointer; pad-row results are discarded.
__device__ __forceinline__ v16h load_a_tile(const float* xrow, int kc, int half) {
    const int o1 = kc * 32 + (half ? 16 : 0);
    const int o2 = kc * 32 + (half ? 24 : 8);
    const v4f a0 = *(const v4f*)(xrow + o1);
    const v4f a1 = *(const v4f*)(xrow + o1 + 4);
    const v4f a2 = *(const v4f*)(xrow + o2);
    const v4f a3 = *(const v4f*)(xrow + o2 + 4);
    v16h A;
    #pragma unroll
    for (int j = 0; j < 4; ++j) {
        A[j]      = (_Float16)a0[j];
        A[4 + j]  = (_Float16)a1[j];
        A[8 + j]  = (_Float16)a2[j];
        A[12 + j] = (_Float16)a3[j];
    }
    return A;
}

// ---------------------------------------------------------------------------
// Kernel 1: fused q/k/v projection (x read once feeds 3 GEMMs), LayerNorm on
// q/k, alpha/sigma heads. Outputs packed f16: q/k row-major, v transposed.
// ---------------------------------------------------------------------------
__global__ __launch_bounds__(32) void proj_ln_kernel(
    const float* __restrict__ x,
    const _Float16* __restrict__ WqP, const _Float16* __restrict__ WkP,
    const _Float16* __restrict__ WvP,
    const float* __restrict__ gq, const float* __restrict__ bq,
    const float* __restrict__ gk, const float* __restrict__ bk,
    const float* __restrict__ Wsig, const float* __restrict__ bsig,
    const float* __restrict__ Wa,   const float* __restrict__ ba,
    _Float16* __restrict__ qh, _Float16* __restrict__ kh, _Float16* __restrict__ vT,
    float* __restrict__ alpha_o, float* __restrict__ sx_o, float* __restrict__ sy_o)
{
    const int lane = threadIdx.x;
    const int half = lane >> 4;
    const int l16  = lane & 15;
    const int b    = blockIdx.x / NTILE;
    const int t0   = (blockIdx.x % NTILE) * 16;

    // A row = lane%16; clamp pad rows to row 0 (their outputs are discarded).
    const int   at    = t0 + l16;
    const float* xrow = x + ((size_t)b * T_TOK + (at < T_TOK ? at : 0)) * DMODEL;

    v8f C[3][4] = {};   // [proj][n-tile], 16x64 each

    const _Float16* WPs[3] = {WqP, WkP, WvP};

    v16h A = load_a_tile(xrow, 0, half);
    for (int kc = 0; kc < DMODEL / 32; ++kc) {
        // Prefetch next chunk's A tile: loads+cvt overlap the 12 WMMAs below.
        v16h Anext = (kc + 1 < DMODEL / 32) ? load_a_tile(xrow, kc + 1, half) : A;
        #pragma unroll
        for (int p = 0; p < 3; ++p) {
            // Batch all 4 B tiles (8x b128 in one clause), then 4 WMMAs.
            const _Float16* wp = WPs[p] + ((size_t)(kc * 4) * 32 + lane) * 16;
            const v16h B0 = *(const v16h*)(wp);
            const v16h B1 = *(const v16h*)(wp + 512);     // 32 lanes * 16 halfs
            const v16h B2 = *(const v16h*)(wp + 1024);
            const v16h B3 = *(const v16h*)(wp + 1536);
            C[p][0] = __builtin_amdgcn_wmma_f32_16x16x32_f16(false, A, false, B0, (short)0, C[p][0], false, false);
            C[p][1] = __builtin_amdgcn_wmma_f32_16x16x32_f16(false, A, false, B1, (short)0, C[p][1], false, false);
            C[p][2] = __builtin_amdgcn_wmma_f32_16x16x32_f16(false, A, false, B2, (short)0, C[p][2], false, false);
            C[p][3] = __builtin_amdgcn_wmma_f32_16x16x32_f16(false, A, false, B3, (short)0, C[p][3], false, false);
        }
        A = Anext;
    }

    // LayerNorm on q (p=0) and k (p=1); C layout: row M=r+half*8, dh=nt*16+l16
    const float* gam[2] = {gq, gk};
    const float* bet[2] = {bq, bk};
    #pragma unroll
    for (int p = 0; p < 2; ++p) {
        #pragma unroll
        for (int r = 0; r < 8; ++r) {
            float s = 0.f, s2 = 0.f;
            #pragma unroll
            for (int nt = 0; nt < 4; ++nt) { float u = C[p][nt][r]; s += u; s2 += u * u; }
            s  = red_sum16(s);
            s2 = red_sum16(s2);
            const float mu  = s * (1.f / 64.f);
            const float var = s2 * (1.f / 64.f) - mu * mu;
            const float inv = rsqrtf(var + 1e-5f);
            #pragma unroll
            for (int nt = 0; nt < 4; ++nt) {
                const int dh = nt * 16 + l16;
                C[p][nt][r] = (C[p][nt][r] - mu) * inv * gam[p][dh] + bet[p][dh];
            }
        }
    }

    // Store packed f16 outputs (pads pre-zeroed by memset).
    #pragma unroll
    for (int r = 0; r < 8; ++r) {
        const int t = t0 + r + half * 8;
        if (t < T_TOK) {
            const size_t qb  = ((size_t)b * TPAD + t) * DH;
            const size_t kb2 = ((size_t)b * KPAD + t) * DH;
            #pragma unroll
            for (int nt = 0; nt < 4; ++nt) {
                const int dh = nt * 16 + l16;
                qh[qb + dh]  = (_Float16)C[0][nt][r];
                kh[kb2 + dh] = (_Float16)C[1][nt][r];
                vT[((size_t)b * DH + dh) * KPAD + t] = (_Float16)C[2][nt][r];
            }
        }
    }

    // alpha = softplus(q@Wa+ba), sig = sigmoid(q@Wsig+bsig) for patch rows
    const float ba0 = ba[0], bs0 = bsig[0], bs1 = bsig[1];
    #pragma unroll
    for (int r = 0; r < 8; ++r) {
        const int t = t0 + r + half * 8;
        float pa = 0.f, p0 = 0.f, p1 = 0.f;
        #pragma unroll
        for (int nt = 0; nt < 4; ++nt) {
            const int dh = nt * 16 + l16;
            const float qv = C[0][nt][r];
            pa += qv * Wa[dh];
            p0 += qv * Wsig[dh * 2 + 0];
            p1 += qv * Wsig[dh * 2 + 1];
        }
        pa = red_sum16(pa); p0 = red_sum16(p0); p1 = red_sum16(p1);
        if (l16 == 0 && t >= 1 && t < T_TOK) {
            const size_t idx = (size_t)b * NPATCH + (t - 1);
            const float aL = pa + ba0;
            alpha_o[idx] = (aL > 20.f) ? aL : log1pf(__expf(aL));   // softplus
            sx_o[idx] = 1.f / (1.f + __expf(-(p0 + bs0)));
            sy_o[idx] = 1.f / (1.f + __expf(-(p1 + bs1)));
        }
    }
}

// ---------------------------------------------------------------------------
// Kernel 2: flash attention with Gaussian spatial bias. One wave = 16 queries.
// V-tile loads are issued before the softmax work so they hide under VALU.
// ---------------------------------------------------------------------------
__global__ __launch_bounds__(32) void attn_kernel(
    const _Float16* __restrict__ qh, const _Float16* __restrict__ kh,
    const _Float16* __restrict__ vT,
    const float* __restrict__ alpha, const float* __restrict__ sx,
    const float* __restrict__ sy,
    float* __restrict__ out)
{
    __shared__ _Float16 Pl[16 * 32];   // probs tile: C-layout -> A-layout bounce

    const int lane = threadIdx.x;
    const int half = lane >> 4;
    const int l16  = lane & 15;
    const int b    = blockIdx.x / NTILE;
    const int q0   = (blockIdx.x % NTILE) * 16;

    const _Float16* qb  = qh + (size_t)b * TPAD * DH;
    const _Float16* kbp = kh + (size_t)b * KPAD * DH;
    const _Float16* vb  = vT + (size_t)b * DH * KPAD;

    // q tile as two A tiles (dh 0-31, 32-63); pad rows are zero.
    v16h Aq[2];
    {
        const _Float16* qr = qb + (size_t)(q0 + l16) * DH;
        #pragma unroll
        for (int c = 0; c < 2; ++c) {
            const v8h a0 = *(const v8h*)(qr + c * 32 + (half ? 16 : 0));
            const v8h a1 = *(const v8h*)(qr + c * 32 + (half ? 24 : 8));
            Aq[c] = cat8(a0, a1);
        }
    }

    // Per-row Gaussian-bias params (this lane's 8 rows)
    float al8[8], i2x[8], i2y[8];
    int   iq8[8], jq8[8];
    #pragma unroll
    for (int r = 0; r < 8; ++r) {
        const int  t  = q0 + r + half * 8;
        const bool pv = (t >= 1 && t < T_TOK);
        const size_t idx = (size_t)b * NPATCH + (pv ? (t - 1) : 0);
        const float a   = pv ? alpha[idx] : 0.f;
        const float sxv = pv ? sx[idx] : 1.f;
        const float syv = pv ? sy[idx] : 1.f;
        al8[r] = a;
        i2x[r] = 0.5f / (sxv * sxv);
        i2y[r] = 0.5f / (syv * syv);
        const int p = pv ? (t - 1) : 0;
        iq8[r] = p >> 5;  jq8[r] = p & 31;
    }

    float m8[8], l8[8];
    v8f   O[4] = {};
    #pragma unroll
    for (int r = 0; r < 8; ++r) { m8[r] = -1e30f; l8[r] = 0.f; }

    for (int kc = 0; kc < NKC; ++kc) {
        const int kb = kc * 32;

        // ---- batch all 4 k B-tiles, then 4 sim WMMAs (staggered waits) ----
        const _Float16* kr0 = kbp + (size_t)(kb + l16) * DH        + (half ? 16 : 0);
        const _Float16* kr1 = kbp + (size_t)(kb + 16 + l16) * DH   + (half ? 16 : 0);
        if (kc + 1 < NKC) __builtin_prefetch(kr0 + 32 * DH, 0, 0);
        const v16h Bk00 = *(const v16h*)(kr0);        // s=0, dh 0-31
        const v16h Bk01 = *(const v16h*)(kr0 + 32);   // s=0, dh 32-63
        const v16h Bk10 = *(const v16h*)(kr1);        // s=1, dh 0-31
        const v16h Bk11 = *(const v16h*)(kr1 + 32);   // s=1, dh 32-63
        v8f Sc[2] = {};
        Sc[0] = __builtin_amdgcn_wmma_f32_16x16x32_f16(false, Aq[0], false, Bk00, (short)0, Sc[0], false, false);
        Sc[0] = __builtin_amdgcn_wmma_f32_16x16x32_f16(false, Aq[1], false, Bk01, (short)0, Sc[0], false, false);
        Sc[1] = __builtin_amdgcn_wmma_f32_16x16x32_f16(false, Aq[0], false, Bk10, (short)0, Sc[1], false, false);
        Sc[1] = __builtin_amdgcn_wmma_f32_16x16x32_f16(false, Aq[1], false, Bk11, (short)0, Sc[1], false, false);

        // ---- issue V B-tiles now: they fly under the softmax VALU work ----
        v16h Bv[4];
        #pragma unroll
        for (int nt = 0; nt < 4; ++nt) {
            const int dh = nt * 16 + l16;
            const _Float16* vr = vb + (size_t)dh * KPAD + kb + (half ? 16 : 0);
            if (nt == 0 && kc + 1 < NKC) __builtin_prefetch(vr + 32, 0, 0);
            Bv[nt] = *(const v16h*)vr;
        }

        // ---- Gaussian bias + key mask + 1/SCALE ----
        float z[2][8];
        #pragma unroll
        for (int s = 0; s < 2; ++s) {
            const int  key    = kb + s * 16 + l16;
            const bool kvalid = (key < T_TOK);
            const int  kp = key - 1;
            const int  ik = kp >> 5, jk = kp & 31;
            #pragma unroll
            for (int r = 0; r < 8; ++r) {
                float sv = Sc[s][r];
                const int t = q0 + r + half * 8;
                if (kvalid && key >= 1 && t >= 1 && t < T_TOK) {
                    const float dx = (float)(jq8[r] - jk);
                    const float dy = (float)(iq8[r] - ik);
                    sv += al8[r] * __expf(-(dx * dx * i2x[r] + dy * dy * i2y[r]));
                }
                z[s][r] = kvalid ? sv * SCALE_INV : -1e30f;
            }
        }

        // ---- online softmax (row stats shared across each 16-lane half) ----
        float cf[8];
        #pragma unroll
        for (int r = 0; r < 8; ++r) {
            float mx = red_max16(fmaxf(z[0][r], z[1][r]));
            const float mn = fmaxf(m8[r], mx);
            const float c0 = __expf(m8[r] - mn);
            const float p0 = __expf(z[0][r] - mn);
            const float p1 = __expf(z[1][r] - mn);
            l8[r] = l8[r] * c0 + red_sum16(p0 + p1);
            m8[r] = mn;
            cf[r] = c0;
            z[0][r] = p0;  z[1][r] = p1;
        }
        #pragma unroll
        for (int nt = 0; nt < 4; ++nt)
            #pragma unroll
            for (int r = 0; r < 8; ++r)
                O[nt][r] *= cf[r];

        // ---- P: C-layout -> LDS (16x32 f16) -> A-layout ----
        #pragma unroll
        for (int s = 0; s < 2; ++s)
            #pragma unroll
            for (int r = 0; r < 8; ++r)
                Pl[(r + half * 8) * 32 + s * 16 + l16] = (_Float16)z[s][r];
        asm volatile("s_wait_dscnt 0" ::: "memory");   // single wave: DS fence only

        v16h Ap;
        {
            const _Float16* pr = &Pl[l16 * 32];
            const v8h p0 = *(const v8h*)(pr + (half ? 16 : 0));
            const v8h p1 = *(const v8h*)(pr + (half ? 24 : 8));
            Ap = cat8(p0, p1);
        }

        // ---- O += P @ V  (K=32 keys; Bv already resident) ----
        O[0] = __builtin_amdgcn_wmma_f32_16x16x32_f16(false, Ap, false, Bv[0], (short)0, O[0], false, false);
        O[1] = __builtin_amdgcn_wmma_f32_16x16x32_f16(false, Ap, false, Bv[1], (short)0, O[1], false, false);
        O[2] = __builtin_amdgcn_wmma_f32_16x16x32_f16(false, Ap, false, Bv[2], (short)0, O[2], false, false);
        O[3] = __builtin_amdgcn_wmma_f32_16x16x32_f16(false, Ap, false, Bv[3], (short)0, O[3], false, false);
    }

    // ---- epilogue: normalize and store ----
    #pragma unroll
    for (int r = 0; r < 8; ++r) {
        const int t = q0 + r + half * 8;
        if (t < T_TOK) {
            const float inv = 1.f / l8[r];
            const size_t base = ((size_t)b * T_TOK + t) * DH;
            #pragma unroll
            for (int nt = 0; nt < 4; ++nt)
                out[base + nt * 16 + l16] = O[nt][r] * inv;
        }
    }
}

// ---------------------------------------------------------------------------
extern "C" void kernel_launch(void* const* d_in, const int* in_sizes, int n_in,
                              void* d_out, int out_size, void* d_ws, size_t ws_size,
                              hipStream_t stream) {
    const float* x    = (const float*)d_in[0];
    const float* Wq   = (const float*)d_in[1];
    const float* Wk   = (const float*)d_in[2];
    const float* Wv   = (const float*)d_in[3];
    const float* gq   = (const float*)d_in[4];
    const float* bq   = (const float*)d_in[5];
    const float* gk   = (const float*)d_in[6];
    const float* bk   = (const float*)d_in[7];
    const float* Wsig = (const float*)d_in[8];
    const float* bsig = (const float*)d_in[9];
    const float* Wa   = (const float*)d_in[10];
    const float* ba   = (const float*)d_in[11];

    const int B = 32;
    const size_t QH  = (size_t)B * TPAD * DH;      // f16 elements
    const size_t KH  = (size_t)B * KPAD * DH;
    const size_t VTT = (size_t)B * DH * KPAD;
    const size_t WPN = (size_t)DMODEL * DH;        // 49152 f16 per W
    const size_t PAT = (size_t)B * NPATCH;

    _Float16* qh  = (_Float16*)d_ws;
    _Float16* kh  = qh + QH;
    _Float16* vT  = kh + KH;
    _Float16* WqP = vT + VTT;
    _Float16* WkP = WqP + WPN;
    _Float16* WvP = WkP + WPN;
    float*    al  = (float*)(WvP + WPN);
    float*    sxw = al + PAT;
    float*    syw = sxw + PAT;

    // Zero q/k/vT pads once so hot loads need no masking (capture-safe).
    hipMemsetAsync(d_ws, 0, (QH + KH + VTT) * sizeof(_Float16), stream);

    pack_w_kernel<<<dim3(3 * 96), dim3(32), 0, stream>>>(Wq, Wk, Wv, WqP, WkP, WvP);

    proj_ln_kernel<<<dim3(B * NTILE), dim3(32), 0, stream>>>(
        x, WqP, WkP, WvP, gq, bq, gk, bk, Wsig, bsig, Wa, ba,
        qh, kh, vT, al, sxw, syw);

    attn_kernel<<<dim3(B * NTILE), dim3(32), 0, stream>>>(
        qh, kh, vT, al, sxw, syw, (float*)d_out);
}
